// DecoderRNN_455266533450
// MI455X (gfx1250) — compile-verified, matrix operand order checked by reading, not run on hardware
//
#include <hip/hip_runtime.h>
#include <stddef.h>
#include <stdint.h>

// ---------------------------------------------------------------------------
// Types
// ---------------------------------------------------------------------------
typedef __bf16 bf16;
typedef __attribute__((ext_vector_type(16))) __bf16 v16bf;
typedef __attribute__((ext_vector_type(8)))  __bf16 v8bf;
typedef __attribute__((ext_vector_type(4)))  __bf16 v4bf;
typedef __attribute__((ext_vector_type(8)))  float  v8f;
typedef __attribute__((ext_vector_type(4)))  float  v4f;

// Problem sizes (fixed by the reference)
#define B_   8192
#define H_   1024
#define V_   1024
#define D_   128
#define IN_  1280      // H + 2D
#define G4H  4096      // 4*H
#define XY2  256       // 16*16

#define LDSP 40        // LDS row pitch in bf16 (32 + 8 pad -> conflict-free)

__device__ __forceinline__ float dsig(float x) { return 1.0f / (1.0f + __expf(-x)); }

// Load one 16x32 (row x k) bf16 fragment (A- or B-operand) from a row-major LDS tile.
// Lane L (L<16): row=L,   elements 0..7 = K 0..7,  8..15 = K 16..23
// Lane L (>=16): row=L-16,elements 0..7 = K 8..15, 8..15 = K 24..31
__device__ __forceinline__ v16bf lds_frag(const bf16* p) {
  v8bf lo = *(const v8bf*)(p);
  v8bf hi = *(const v8bf*)(p + 16);
  return __builtin_shufflevector(lo, hi, 0,1,2,3,4,5,6,7,8,9,10,11,12,13,14,15);
}

// ---------------------------------------------------------------------------
// Generic WMMA GEMM:  out[M,N] = act( sum_s A_s[M,Ks](bf16) @ W_s[N,Ks]^T(f32) + bias )
// Block tile 128x128, 256 threads (8 wave32), wave grid 4(M) x 2(N),
// per-wave 32x64 = 2x4 fragments of 16x16, K-step 32, LDS double-buffered.
// ---------------------------------------------------------------------------
__global__ __launch_bounds__(256)
void gemm_bf16_wmma(const bf16* A0, int lda0, const float* W0, int ldw0, int K0,
                    const bf16* A1, int lda1, const float* W1, int ldw1, int K1,
                    const bf16* A2, int lda2, const float* W2, int ldw2, int K2,
                    const float* bias, int act,
                    float* outF, int ldoF, bf16* outB, int ldoB)
{
  __shared__ bf16 sA[2][128 * LDSP];
  __shared__ bf16 sW[2][128 * LDSP];

  const int t     = threadIdx.x;
  const int wid   = t >> 5;
  const int lane  = t & 31;
  const int lrow  = lane & 15;
  const int lhalf = lane >> 4;
  const int waveM = wid >> 1;          // 0..3
  const int waveN = wid & 1;           // 0..1
  const int m0    = blockIdx.y * 128;
  const int n0    = blockIdx.x * 128;
  const int kbase = lhalf * 8;

  v8f acc[2][4];
#pragma unroll
  for (int mi = 0; mi < 2; ++mi)
#pragma unroll
    for (int ni = 0; ni < 4; ++ni) {
      v8f z = {0.f,0.f,0.f,0.f,0.f,0.f,0.f,0.f};
      acc[mi][ni] = z;
    }

  // Per-thread global->LDS chunk coordinates.
  // A tile: 128 rows x 32 k bf16 = 512 chunks of 8 bf16 (16B); 2 per thread.
  int aRow[2], aCol[2];
#pragma unroll
  for (int i = 0; i < 2; ++i) { int id = t + i * 256; aRow[i] = id >> 2; aCol[i] = (id & 3) * 8; }
  // W tile: 128 n-rows x 32 k f32 = 1024 chunks of float4; 4 per thread.
  int wRow[4], wCol[4];
#pragma unroll
  for (int i = 0; i < 4; ++i) { int id = t + i * 256; wRow[i] = id >> 3; wCol[i] = (id & 7) * 4; }

  const bf16*  segA[3]  = {A0, A1, A2};
  const float* segW[3]  = {W0, W1, W2};
  const int    segLA[3] = {lda0, lda1, lda2};
  const int    segLW[3] = {ldw0, ldw1, ldw2};
  const int    segK[3]  = {K0, K1, K2};

  for (int s = 0; s < 3; ++s) {
    const int K = segK[s];
    if (K <= 0) continue;
    const bf16*  A   = segA[s];
    const float* W   = segW[s];
    const int    lda = segLA[s];
    const int    ldw = segLW[s];
    const int    nk  = K >> 5;

    v8bf aReg[2];
    v4f  wReg[4];

    // Prologue: fetch K-block 0 into registers.
#pragma unroll
    for (int i = 0; i < 2; ++i)
      aReg[i] = *(const v8bf*)(A + (size_t)(m0 + aRow[i]) * lda + aCol[i]);
#pragma unroll
    for (int i = 0; i < 4; ++i)
      wReg[i] = *(const v4f*)(W + (size_t)(n0 + wRow[i]) * ldw + wCol[i]);

    __syncthreads();   // buffers free from previous segment's compute

#pragma unroll
    for (int i = 0; i < 2; ++i)
      *(v8bf*)(&sA[0][aRow[i] * LDSP + aCol[i]]) = aReg[i];
#pragma unroll
    for (int i = 0; i < 4; ++i) {
      v4bf c;
#pragma unroll
      for (int j = 0; j < 4; ++j) c[j] = (bf16)wReg[i][j];
      *(v4bf*)(&sW[0][wRow[i] * LDSP + wCol[i]]) = c;
    }

    for (int kb = 0; kb < nk; ++kb) {
      const int cur = kb & 1;
      const int nxt = cur ^ 1;

      if (kb + 1 < nk) {                       // prefetch next K-block
        const int kpos = (kb + 1) << 5;
#pragma unroll
        for (int i = 0; i < 2; ++i)
          aReg[i] = *(const v8bf*)(A + (size_t)(m0 + aRow[i]) * lda + kpos + aCol[i]);
#pragma unroll
        for (int i = 0; i < 4; ++i)
          wReg[i] = *(const v4f*)(W + (size_t)(n0 + wRow[i]) * ldw + kpos + wCol[i]);
      }

      __syncthreads();                         // sA/sW[cur] ready

      v16bf af[2], bfr[4];
#pragma unroll
      for (int mi = 0; mi < 2; ++mi)
        af[mi] = lds_frag(&sA[cur][(waveM * 32 + mi * 16 + lrow) * LDSP + kbase]);
#pragma unroll
      for (int ni = 0; ni < 4; ++ni)
        bfr[ni] = lds_frag(&sW[cur][(waveN * 64 + ni * 16 + lrow) * LDSP + kbase]);

#pragma unroll
      for (int mi = 0; mi < 2; ++mi)
#pragma unroll
        for (int ni = 0; ni < 4; ++ni)
          acc[mi][ni] = __builtin_amdgcn_wmma_f32_16x16x32_bf16(
              false, af[mi], false, bfr[ni], (short)0, acc[mi][ni], false, false);

      if (kb + 1 < nk) {                       // stage prefetch into other buffer
#pragma unroll
        for (int i = 0; i < 2; ++i)
          *(v8bf*)(&sA[nxt][aRow[i] * LDSP + aCol[i]]) = aReg[i];
#pragma unroll
        for (int i = 0; i < 4; ++i) {
          v4bf c;
#pragma unroll
          for (int j = 0; j < 4; ++j) c[j] = (bf16)wReg[i][j];
          *(v4bf*)(&sW[nxt][wRow[i] * LDSP + wCol[i]]) = c;
        }
      }
    }
  }

  // Epilogue: C/D layout -> VGPR r holds row (r) for lanes 0-15, (r+8) for 16-31; col = lane%16.
#pragma unroll
  for (int ni = 0; ni < 4; ++ni) {
    const int col = n0 + waveN * 64 + ni * 16 + lrow;
    const float bv = bias ? bias[col] : 0.0f;
#pragma unroll
    for (int mi = 0; mi < 2; ++mi) {
      const int rowb = m0 + waveM * 32 + mi * 16 + lhalf * 8;
#pragma unroll
      for (int r = 0; r < 8; ++r) {
        float v = acc[mi][ni][r] + bv;
        if (act == 1)      v = fmaxf(v, 0.0f);
        else if (act == 2) v = dsig(v);
        const size_t row = (size_t)(rowb + r);
        if (outF) outF[row * (size_t)ldoF + col] = v;
        if (outB) outB[row * (size_t)ldoB + col] = (bf16)v;
      }
    }
  }
}

// ---------------------------------------------------------------------------
// Pointwise / prep kernels
// ---------------------------------------------------------------------------
__global__ void k_xywh(const float* tx, const float* ty, const float* tw, const float* th,
                       const float* xyW, const float* xyb, const float* whW, const float* whb,
                       bf16* rnn_in)
{
  int i = blockIdx.x * blockDim.x + threadIdx.x;   // B*128
  int b = i >> 7, d = i & 127;
  float xv = tx[b] * xyW[2 * d] + ty[b] * xyW[2 * d + 1] + xyb[d];
  float wv = tw[b] * whW[2 * d] + th[b] * whW[2 * d + 1] + whb[d];
  rnn_in[(size_t)b * IN_ + d]       = (bf16)xv;
  rnn_in[(size_t)b * IN_ + 128 + d] = (bf16)wv;
}

__global__ void k_emb(const int* tl, const float* emb, bf16* rnn_in)
{
  int i = blockIdx.x * blockDim.x + threadIdx.x;   // B*H
  int b = i >> 10, h = i & 1023;
  int l = tl[b];
  rnn_in[(size_t)b * IN_ + 256 + h] = (bf16)emb[(size_t)l * H_ + h];
}

__global__ void k_cvt(const float* in, bf16* out)
{
  int i = blockIdx.x * blockDim.x + threadIdx.x;
  out[i] = (bf16)in[i];
}

__global__ void k_nxy(const float* nxy, const float* W, const float* b, bf16* out)
{
  int i = blockIdx.x * blockDim.x + threadIdx.x;   // B*128
  int bb = i >> 7, d = i & 127;
  float v = nxy[2 * bb] * W[2 * d] + nxy[2 * bb + 1] * W[2 * d + 1] + b[d];
  out[(size_t)bb * D_ + d] = (bf16)v;
}

__global__ void k_bsum(const float* b0, const float* b1, float* out)
{
  int i = blockIdx.x * blockDim.x + threadIdx.x;   // 4096
  out[i] = b0[i] + b1[i];
}

__global__ void k_lstm(const float* gates, const float* c0,
                       float* outH, float* outC, bf16* hbf)
{
  int i = blockIdx.x * blockDim.x + threadIdx.x;   // B*H
  int b = i >> 10, h = i & 1023;
  const float* g = gates + (size_t)b * G4H;
  float gi = g[h], gf = g[1024 + h], gg = g[2048 + h], go = g[3072 + h];
  float c  = dsig(gf) * c0[i] + dsig(gi) * tanhf(gg);
  float hv = dsig(go) * tanhf(c);
  outH[i] = hv;
  outC[i] = c;
  hbf[i]  = (bf16)hv;
}

__global__ void k_wh2(const bf16* hidden2, const float* W, const float* b2, float* out)
{
  int i = blockIdx.x * blockDim.x + threadIdx.x;   // B*2
  int bb = i >> 1, n = i & 1;
  const bf16*  hd = hidden2 + (size_t)bb * 512;
  const float* w  = W + (size_t)n * 512;
  float acc = b2[n];
#pragma unroll 8
  for (int k = 0; k < 512; ++k) acc += (float)hd[k] * w[k];
  out[i] = dsig(acc);
}

// ---------------------------------------------------------------------------
// Host launcher
// ---------------------------------------------------------------------------
static inline void launch_gemm(hipStream_t s, int N,
    const bf16* A0, int lda0, const float* W0, int ldw0, int K0,
    const bf16* A1, int lda1, const float* W1, int ldw1, int K1,
    const bf16* A2, int lda2, const float* W2, int ldw2, int K2,
    const float* bias, int act, float* oF, int ldoF, bf16* oB, int ldoB)
{
  dim3 grid(N / 128, B_ / 128);
  gemm_bf16_wmma<<<grid, 256, 0, s>>>(A0, lda0, W0, ldw0, K0,
                                      A1, lda1, W1, ldw1, K1,
                                      A2, lda2, W2, ldw2, K2,
                                      bias, act, oF, ldoF, oB, ldoB);
}

extern "C" void kernel_launch(void* const* d_in, const int* in_sizes, int n_in,
                              void* d_out, int out_size, void* d_ws, size_t ws_size,
                              hipStream_t stream)
{
  (void)in_sizes; (void)n_in; (void)out_size; (void)ws_size;

  // Inputs (setup_inputs dict order; params dict flattened in insertion order)
  const int*   target_l = (const int*)  d_in[0];
  const float* target_x = (const float*)d_in[1];
  const float* target_y = (const float*)d_in[2];
  const float* target_w = (const float*)d_in[3];
  const float* target_h = (const float*)d_in[4];
  const float* h0       = (const float*)d_in[5];   // (1,B,H)
  const float* c0       = (const float*)d_in[6];   // (1,B,H)
  /* d_in[7] = encoder_output, unused (no attention) */
  const float* next_l   = (const float*)d_in[8];   // (B,V)
  const float* next_xy  = (const float*)d_in[9];   // (B,2)
  const float* embedding= (const float*)d_in[10];  // (V,H)
  const float* xy_W     = (const float*)d_in[11];
  const float* xy_b     = (const float*)d_in[12];
  const float* wh_W     = (const float*)d_in[13];
  const float* wh_b     = (const float*)d_in[14];
  const float* W_ih     = (const float*)d_in[15];  // (4H, IN)
  const float* b_ih     = (const float*)d_in[16];
  const float* W_hh     = (const float*)d_in[17];  // (4H, H)
  const float* b_hh     = (const float*)d_in[18];
  const float* class_W  = (const float*)d_in[19];  // (V, 2H)
  const float* class_b  = (const float*)d_in[20];
  const float* mxy_W1   = (const float*)d_in[21];  // (512, 2H)
  const float* mxy_b1   = (const float*)d_in[22];
  const float* mxy_W2   = (const float*)d_in[23];  // (256, 512)
  const float* mxy_b2   = (const float*)d_in[24];
  const float* nxy_W    = (const float*)d_in[25];  // (D, 2)
  const float* nxy_b    = (const float*)d_in[26];
  const float* mwh_W1   = (const float*)d_in[27];  // (512, 2176)
  const float* mwh_b1   = (const float*)d_in[28];
  const float* mwh_W2   = (const float*)d_in[29];  // (2, 512)
  const float* mwh_b2   = (const float*)d_in[30];

  // Workspace layout (256B aligned slices)
  char*  ws  = (char*)d_ws;
  size_t off = 0;
  auto alloc = [&](size_t bytes) -> void* {
    void* p = ws + off;
    off += (bytes + 255) & ~(size_t)255;
    return p;
  };
  float* gates   = (float*)alloc((size_t)B_ * G4H * sizeof(float));   // 134 MB
  bf16*  rnn_in  = (bf16*) alloc((size_t)B_ * IN_ * sizeof(bf16));    // [xy|wh|emb]
  bf16*  h0bf    = (bf16*) alloc((size_t)B_ * H_  * sizeof(bf16));
  bf16*  nlbf    = (bf16*) alloc((size_t)B_ * V_  * sizeof(bf16));
  bf16*  hbf     = (bf16*) alloc((size_t)B_ * H_  * sizeof(bf16));
  bf16*  nxybf   = (bf16*) alloc((size_t)B_ * D_  * sizeof(bf16));
  bf16*  hidden1 = (bf16*) alloc((size_t)B_ * 512 * sizeof(bf16));
  bf16*  hidden2 = (bf16*) alloc((size_t)B_ * 512 * sizeof(bf16));
  float* bsum    = (float*)alloc((size_t)G4H * sizeof(float));

  // Output slices (tuple concatenated flat, all f32)
  float* outClass = (float*)d_out;                       // (B,V)
  float* outXY    = outClass + (size_t)B_ * V_;          // (B,256)
  float* outWH    = outXY    + (size_t)B_ * XY2;         // (B,2)
  float* outH     = outWH    + (size_t)B_ * 2;           // (1,B,H)
  float* outC     = outH     + (size_t)B_ * H_;          // (1,B,H)

  // --- Prep (independent) ---
  k_bsum<<<G4H / 256, 256, 0, stream>>>(b_ih, b_hh, bsum);
  k_xywh<<<(B_ * 128) / 256, 256, 0, stream>>>(target_x, target_y, target_w, target_h,
                                               xy_W, xy_b, wh_W, wh_b, rnn_in);
  k_emb<<<(B_ * H_) / 256, 256, 0, stream>>>(target_l, embedding, rnn_in);
  k_cvt<<<(B_ * H_) / 256, 256, 0, stream>>>(h0, h0bf);
  k_cvt<<<(B_ * V_) / 256, 256, 0, stream>>>(next_l, nlbf);
  k_nxy<<<(B_ * 128) / 256, 256, 0, stream>>>(next_xy, nxy_W, nxy_b, nxybf);

  // --- gates = rnn_in @ W_ih^T + h0 @ W_hh^T + (b_ih+b_hh) ---
  launch_gemm(stream, G4H,
              rnn_in, IN_, W_ih, IN_, IN_,
              h0bf,   H_,  W_hh, H_,  H_,
              nullptr, 0, nullptr, 0, 0,
              bsum, /*act*/0, gates, G4H, nullptr, 0);

  // --- LSTM pointwise: h, c ---
  k_lstm<<<(B_ * H_) / 256, 256, 0, stream>>>(gates, c0, outH, outC, hbf);

  // --- class = [h, emb] @ class_W^T + class_b ---
  launch_gemm(stream, V_,
              hbf,           H_,  class_W,        2 * H_, H_,
              rnn_in + 256,  IN_, class_W + H_,   2 * H_, H_,
              nullptr, 0, nullptr, 0, 0,
              class_b, /*act*/0, outClass, V_, nullptr, 0);

  // --- hidden1 = relu([h, next_l] @ mxy_W1^T + b1)  (bf16 out) ---
  launch_gemm(stream, 512,
              hbf,  H_, mxy_W1,       2 * H_, H_,
              nlbf, V_, mxy_W1 + H_,  2 * H_, V_,
              nullptr, 0, nullptr, 0, 0,
              mxy_b1, /*act*/1, nullptr, 0, hidden1, 512);

  // --- xy_out = relu(hidden1 @ mxy_W2^T + b2) ---
  launch_gemm(stream, XY2,
              hidden1, 512, mxy_W2, 512, 512,
              nullptr, 0, nullptr, 0, 0,
              nullptr, 0, nullptr, 0, 0,
              mxy_b2, /*act*/1, outXY, XY2, nullptr, 0);

  // --- hidden2 = relu([h, next_l, nxy] @ mwh_W1^T + b1)  (3 segments, bf16 out) ---
  launch_gemm(stream, 512,
              hbf,   H_, mwh_W1,            2176, H_,
              nlbf,  V_, mwh_W1 + H_,       2176, V_,
              nxybf, D_, mwh_W1 + H_ + V_,  2176, D_,
              mwh_b1, /*act*/1, nullptr, 0, hidden2, 512);

  // --- wh_out = sigmoid(hidden2 @ mwh_W2^T + b2)  (N=2, trivial) ---
  k_wh2<<<(B_ * 2) / 256, 256, 0, stream>>>(hidden2, mwh_W2, mwh_b2, outWH);
}
